// MultiHeadAttention_15822659518621
// MI455X (gfx1250) — matrix operand order repaired and checked by reading, not compiled, not run on hardware
//
#include <hip/hip_runtime.h>
#include <hip/hip_bf16.h>

// ---------------------------------------------------------------------------
// MultiHeadAttention forward for gfx1250 (CDNA5, wave32, WMMA bf16 path)
// ---------------------------------------------------------------------------

typedef __bf16  v16bf __attribute__((ext_vector_type(16)));
typedef float   v8f   __attribute__((ext_vector_type(8)));
typedef float   f4v   __attribute__((ext_vector_type(4)));
typedef int     i4v   __attribute__((ext_vector_type(4)));

union Frag {            // one WMMA A/B operand: 16 bf16 per lane = 32 bytes
    v16bf v;
    i4v   q[2];
};

#define SEQ    1024
#define BATCH  8
#define DMODEL 1024
#define HEADS  16
#define DK     64
#define MROWS  (SEQ * BATCH)     // 8192

// ---------------------------------------------------------------------------
// Generic  Y = A[M,1024] * W[1024,1024]^T  tiled GEMM (MODE is compile-time).
//   MODE 0: Y -> bf16  [b,h][s][dk]      (Q / K staging)
//   MODE 1: Y -> bf16  [b,h][dk][s]      (V staged transposed)
//   MODE 2: Y -> f32   [m][n] + bias     (final output projection)
// Block: 256 thr, tile 128x128, K-step 32. Wave: 32x64 (2x4 WMMA tiles).
// Software-pipelined: next K-slab global loads issued before the WMMA block.
// ---------------------------------------------------------------------------
template <int MODE>
__global__ __launch_bounds__(256) void gemm_xwt_kernel(
    const float* __restrict__ A,        // [8192, 1024]
    const float* __restrict__ W,        // [1024, 1024]
    const float* __restrict__ bias,     // [1024] (MODE 2) or nullptr
    float* __restrict__ outF,           // MODE 2
    __bf16* __restrict__ outBF)         // MODE 0/1
{
    __shared__ __bf16 lA[128 * 40];   // 80 B rows -> 16B aligned chunks
    __shared__ __bf16 lB[128 * 40];

    const int tid  = threadIdx.x;
    const int bm   = blockIdx.y;     // 64 M tiles
    const int bn   = blockIdx.x;     // 8  N tiles
    const int wave = tid >> 5;
    const int lane = tid & 31;
    const int hi   = lane >> 4;
    const int lo   = lane & 15;
    const int wm   = (wave & 3) * 32;
    const int wn   = (wave >> 2) * 64;

    // each thread owns 4 A-chunks + 4 B-chunks of the 128x32 staging tiles
    int rr[4], cc[4];
    #pragma unroll
    for (int i = 0; i < 4; ++i) {
        int c = tid + i * 256;               // 0..1023 chunk id
        rr[i] = c >> 3;
        cc[i] = (c & 7) * 4;
    }

    v8f acc[2][4] = {};
    f4v ra[4], rb[4];

    // prologue: fetch first K-slab
    #pragma unroll
    for (int i = 0; i < 4; ++i) {
        ra[i] = *(const f4v*)(A + (size_t)(bm * 128 + rr[i]) * DMODEL + cc[i]);
        rb[i] = *(const f4v*)(W + (size_t)(bn * 128 + rr[i]) * DMODEL + cc[i]);
    }

    for (int kk = 0; kk < DMODEL; kk += 32) {
        __syncthreads();
        #pragma unroll
        for (int i = 0; i < 4; ++i) {
            __bf16* da = &lA[rr[i] * 40 + cc[i]];
            da[0] = (__bf16)ra[i].x; da[1] = (__bf16)ra[i].y;
            da[2] = (__bf16)ra[i].z; da[3] = (__bf16)ra[i].w;
            __bf16* db = &lB[rr[i] * 40 + cc[i]];
            db[0] = (__bf16)rb[i].x; db[1] = (__bf16)rb[i].y;
            db[2] = (__bf16)rb[i].z; db[3] = (__bf16)rb[i].w;
        }
        __syncthreads();

        // issue next slab's global loads so they overlap the WMMA block
        const int kn = (kk + 32 < DMODEL) ? kk + 32 : kk;
        #pragma unroll
        for (int i = 0; i < 4; ++i) {
            ra[i] = *(const f4v*)(A + (size_t)(bm * 128 + rr[i]) * DMODEL + kn + cc[i]);
            rb[i] = *(const f4v*)(W + (size_t)(bn * 128 + rr[i]) * DMODEL + kn + cc[i]);
        }
        if (kk + 64 < DMODEL) {   // speculative L2 prefetch one slab further
            __builtin_prefetch(A + (size_t)(bm * 128 + rr[0]) * DMODEL + kk + 64 + cc[0], 0, 0);
            __builtin_prefetch(W + (size_t)(bn * 128 + rr[0]) * DMODEL + kk + 64 + cc[0], 0, 0);
        }

        Frag a[2], b[4];
        #pragma unroll
        for (int i = 0; i < 2; ++i) {
            const __bf16* base = &lA[(wm + i * 16 + lo) * 40];
            a[i].q[0] = *(const i4v*)(base + hi * 8);
            a[i].q[1] = *(const i4v*)(base + hi * 8 + 16);
        }
        #pragma unroll
        for (int j = 0; j < 4; ++j) {
            const __bf16* base = &lB[(wn + j * 16 + lo) * 40];
            b[j].q[0] = *(const i4v*)(base + hi * 8);
            b[j].q[1] = *(const i4v*)(base + hi * 8 + 16);
        }
        #pragma unroll
        for (int i = 0; i < 2; ++i)
            #pragma unroll
            for (int j = 0; j < 4; ++j)
                acc[i][j] = __builtin_amdgcn_wmma_f32_16x16x32_bf16(
                    false, a[i].v, false, b[j].v, (short)0, acc[i][j], false, false);
    }

    #pragma unroll
    for (int i = 0; i < 2; ++i)
        #pragma unroll
        for (int j = 0; j < 4; ++j)
            #pragma unroll
            for (int r = 0; r < 8; ++r) {
                int mg = bm * 128 + wm + i * 16 + r + hi * 8;
                int ng = bn * 128 + wn + j * 16 + lo;
                float v = acc[i][j][r];
                if (MODE == 2) {
                    outF[(size_t)mg * DMODEL + ng] = v + bias[ng];
                } else {
                    int s = mg >> 3, bb = mg & 7;          // m = s*B + b
                    int h = ng >> 6, dki = ng & 63;        // n = h*64 + dk
                    size_t idx = (MODE == 0)
                        ? ((size_t)(bb * HEADS + h) * SEQ + s) * DK + dki
                        : ((size_t)(bb * HEADS + h) * DK + dki) * SEQ + s;
                    outBF[idx] = (__bf16)v;
                }
            }
}

// ---------------------------------------------------------------------------
// scores[b,h][q][k] = (Q . K) / 8, masked.  Per (b,h): 1024x1024x64 GEMM.
// ---------------------------------------------------------------------------
__global__ __launch_bounds__(256) void scores_kernel(
    const __bf16* __restrict__ Qbf,   // [b,h][s][dk] bf16
    const __bf16* __restrict__ Kbf,   // [b,h][s][dk] bf16
    const int* __restrict__ mask,     // [B][S][S]
    float* __restrict__ attn)         // [b,h][q][k] f32 (raw scores)
{
    const int bh = blockIdx.z;
    const int b  = bh >> 4;
    const __bf16* Q  = Qbf + (size_t)bh * SEQ * DK;
    const __bf16* Kp = Kbf + (size_t)bh * SEQ * DK;

    __shared__ __bf16 lQ[128 * 72];   // 144 B rows
    __shared__ __bf16 lK[128 * 72];

    const int tid = threadIdx.x;
    const int bm = blockIdx.y, bn = blockIdx.x;
    const int wave = tid >> 5, lane = tid & 31;
    const int hi = lane >> 4, lo = lane & 15;
    const int wm = (wave & 3) * 32, wn = (wave >> 2) * 64;

    // stage 128x32 bf16 tiles (dk 0..31) (512 x 16B chunks each, 2/thread)
    #pragma unroll
    for (int i = 0; i < 2; ++i) {
        int c = tid + i * 256;
        int r = c >> 2, c8 = (c & 3) * 8;
        *(i4v*)&lQ[r * 72 + c8] = *(const i4v*)(Q  + (size_t)(bm * 128 + r) * DK + c8);
        *(i4v*)&lK[r * 72 + c8] = *(const i4v*)(Kp + (size_t)(bn * 128 + r) * DK + c8);
    }
    __syncthreads();

    v8f acc[2][4] = {};
    #pragma unroll
    for (int kk = 0; kk < DK; kk += 32) {
        Frag a[2], bb[4];
        #pragma unroll
        for (int i = 0; i < 2; ++i) {
            const __bf16* base = &lQ[(wm + i * 16 + lo) * 72 + kk];
            a[i].q[0] = *(const i4v*)(base + hi * 8);
            a[i].q[1] = *(const i4v*)(base + hi * 8 + 16);
        }
        #pragma unroll
        for (int j = 0; j < 4; ++j) {
            const __bf16* base = &lK[(wn + j * 16 + lo) * 72 + kk];
            bb[j].q[0] = *(const i4v*)(base + hi * 8);
            bb[j].q[1] = *(const i4v*)(base + hi * 8 + 16);
        }
        #pragma unroll
        for (int i = 0; i < 2; ++i)
            #pragma unroll
            for (int j = 0; j < 4; ++j)
                acc[i][j] = __builtin_amdgcn_wmma_f32_16x16x32_bf16(
                    false, a[i].v, false, bb[j].v, (short)0, acc[i][j], false, false);
    }

    const float inv_scale = 0.125f;   // 1/sqrt(64)
    #pragma unroll
    for (int i = 0; i < 2; ++i)
        #pragma unroll
        for (int j = 0; j < 4; ++j)
            #pragma unroll
            for (int r = 0; r < 8; ++r) {
                int q = bm * 128 + wm + i * 16 + r + hi * 8;
                int k = bn * 128 + wn + j * 16 + lo;
                float v = acc[i][j][r] * inv_scale;
                int mv = mask[(size_t)b * SEQ * SEQ + (size_t)q * SEQ + k];
                if (mv == 0) v = -1e9f;
                attn[((size_t)bh * SEQ + q) * SEQ + k] = v;
            }
}

// ---------------------------------------------------------------------------
// Row softmax in place over the 1024-wide attn rows. One block per row.
// wave32 shuffle reductions + one LDS hop across the 8 waves.
// ---------------------------------------------------------------------------
__global__ __launch_bounds__(256) void softmax_kernel(float* __restrict__ attn)
{
    const size_t row = blockIdx.x;
    float* p = attn + row * SEQ;
    const int tid  = threadIdx.x;
    const int wave = tid >> 5;
    const int lane = tid & 31;
    __shared__ float red[8];

    f4v v = *(f4v*)(p + tid * 4);
    float m = fmaxf(fmaxf(v.x, v.y), fmaxf(v.z, v.w));
    #pragma unroll
    for (int o = 16; o > 0; o >>= 1) m = fmaxf(m, __shfl_xor(m, o, 32));
    if (lane == 0) red[wave] = m;
    __syncthreads();
    m = red[0];
    #pragma unroll
    for (int w = 1; w < 8; ++w) m = fmaxf(m, red[w]);
    __syncthreads();

    v.x = __expf(v.x - m); v.y = __expf(v.y - m);
    v.z = __expf(v.z - m); v.w = __expf(v.w - m);
    float s = v.x + v.y + v.z + v.w;
    #pragma unroll
    for (int o = 16; o > 0; o >>= 1) s += __shfl_xor(s, o, 32);
    if (lane == 0) red[wave] = s;
    __syncthreads();
    s = red[0];
    #pragma unroll
    for (int w = 1; w < 8; ++w) s += red[w];

    float inv = 1.0f / s;
    v.x *= inv; v.y *= inv; v.z *= inv; v.w *= inv;
    *(f4v*)(p + tid * 4) = v;
}

// ---------------------------------------------------------------------------
// ctx[b,h][s][dk] = attn[b,h] (1024x1024, f32->bf16) x V^T[b,h] (64x1024 bf16)
// stored into ctx[(s*B+b)][h*64+dk] f32 for the output projection GEMM.
// Software-pipelined like the projection GEMM.
// ---------------------------------------------------------------------------
__global__ __launch_bounds__(256) void ctx_kernel(
    const float* __restrict__ attn,
    const __bf16* __restrict__ Vt,    // [b,h][dk][s] bf16
    float* __restrict__ ctx)          // [8192][1024] f32
{
    const int bh = blockIdx.y;
    const int b = bh >> 4, h = bh & 15;
    const int bm = blockIdx.x;               // 8 M tiles of 128
    const float* P = attn + (size_t)bh * SEQ * SEQ;
    const __bf16* V = Vt + (size_t)bh * DK * SEQ;

    __shared__ __bf16 lP[128 * 40];
    __shared__ __bf16 lV[64 * 40];

    const int tid = threadIdx.x;
    const int wave = tid >> 5, lane = tid & 31;
    const int hi = lane >> 4, lo = lane & 15;
    const int wm = (wave & 3) * 32, wn = (wave >> 2) * 32;

    int rr[4], cc[4];
    #pragma unroll
    for (int i = 0; i < 4; ++i) {
        int c = tid + i * 256;
        rr[i] = c >> 3;
        cc[i] = (c & 7) * 4;
    }
    const int vr = tid >> 2, vc = (tid & 3) * 8;   // V tile: 256 x 16B chunks

    v8f acc[2][2] = {};
    f4v rp[4];
    i4v rv;

    #pragma unroll
    for (int i = 0; i < 4; ++i)
        rp[i] = *(const f4v*)(P + (size_t)(bm * 128 + rr[i]) * SEQ + cc[i]);
    rv = *(const i4v*)(V + (size_t)vr * SEQ + vc);

    for (int kk = 0; kk < SEQ; kk += 32) {
        __syncthreads();
        #pragma unroll
        for (int i = 0; i < 4; ++i) {
            __bf16* d = &lP[rr[i] * 40 + cc[i]];
            d[0] = (__bf16)rp[i].x; d[1] = (__bf16)rp[i].y;
            d[2] = (__bf16)rp[i].z; d[3] = (__bf16)rp[i].w;
        }
        *(i4v*)&lV[vr * 40 + vc] = rv;
        __syncthreads();

        const int kn = (kk + 32 < SEQ) ? kk + 32 : kk;
        #pragma unroll
        for (int i = 0; i < 4; ++i)
            rp[i] = *(const f4v*)(P + (size_t)(bm * 128 + rr[i]) * SEQ + kn + cc[i]);
        rv = *(const i4v*)(V + (size_t)vr * SEQ + kn + vc);

        Frag a[2], bb[2];
        #pragma unroll
        for (int i = 0; i < 2; ++i) {
            const __bf16* base = &lP[(wm + i * 16 + lo) * 40];
            a[i].q[0] = *(const i4v*)(base + hi * 8);
            a[i].q[1] = *(const i4v*)(base + hi * 8 + 16);
        }
        #pragma unroll
        for (int j = 0; j < 2; ++j) {
            const __bf16* base = &lV[(wn + j * 16 + lo) * 40];
            bb[j].q[0] = *(const i4v*)(base + hi * 8);
            bb[j].q[1] = *(const i4v*)(base + hi * 8 + 16);
        }
        #pragma unroll
        for (int i = 0; i < 2; ++i)
            #pragma unroll
            for (int j = 0; j < 2; ++j)
                acc[i][j] = __builtin_amdgcn_wmma_f32_16x16x32_bf16(
                    false, a[i].v, false, bb[j].v, (short)0, acc[i][j], false, false);
    }

    #pragma unroll
    for (int i = 0; i < 2; ++i)
        #pragma unroll
        for (int j = 0; j < 2; ++j)
            #pragma unroll
            for (int r = 0; r < 8; ++r) {
                int s = bm * 128 + wm + i * 16 + r + hi * 8;
                int n = wn + j * 16 + lo;
                ctx[(size_t)(s * BATCH + b) * DMODEL + h * DK + n] = acc[i][j][r];
            }
}

// ---------------------------------------------------------------------------
extern "C" void kernel_launch(void* const* d_in, const int* in_sizes, int n_in,
                              void* d_out, int out_size, void* d_ws, size_t ws_size,
                              hipStream_t stream)
{
    const float* query = (const float*)d_in[0];
    const float* key_  = (const float*)d_in[1];
    const float* value = (const float*)d_in[2];
    const int*   mask  = (const int*)  d_in[3];
    const float* w_q   = (const float*)d_in[4];
    const float* w_k   = (const float*)d_in[5];
    const float* w_v   = (const float*)d_in[6];
    const float* w_o   = (const float*)d_in[7];
    const float* b_o   = (const float*)d_in[8];

    float* out  = (float*)d_out;
    float* attn = out + (size_t)SEQ * BATCH * DMODEL;    // second output

    const size_t NQKV = (size_t)MROWS * DMODEL;          // 8388608 elements
    __bf16* qbf  = (__bf16*)d_ws;
    __bf16* kbf  = qbf  + NQKV;
    __bf16* vtbf = kbf  + NQKV;
    float*  ctx  = (float*)(vtbf + NQKV);

    dim3 blk(256);
    dim3 gProj(DMODEL / 128, MROWS / 128);               // (8, 64)

    gemm_xwt_kernel<0><<<gProj, blk, 0, stream>>>(query, w_q, nullptr, nullptr, qbf);
    gemm_xwt_kernel<0><<<gProj, blk, 0, stream>>>(key_,  w_k, nullptr, nullptr, kbf);
    gemm_xwt_kernel<1><<<gProj, blk, 0, stream>>>(value, w_v, nullptr, nullptr, vtbf);

    scores_kernel<<<dim3(SEQ / 128, SEQ / 128, BATCH * HEADS), blk, 0, stream>>>(
        qbf, kbf, mask, attn);

    softmax_kernel<<<dim3(BATCH * HEADS * SEQ), blk, 0, stream>>>(attn);

    ctx_kernel<<<dim3(SEQ / 128, BATCH * HEADS), blk, 0, stream>>>(attn, vtbf, ctx);

    gemm_xwt_kernel<2><<<gProj, blk, 0, stream>>>(ctx, w_o, b_o, out, nullptr);
}
